// MOE_28698971472141
// MI455X (gfx1250) — matrix operand order repaired and checked by reading, not compiled, not run on hardware
//
#include <hip/hip_runtime.h>
#include <math.h>

// MoE forward for MI455X (gfx1250, wave32, WMMA).
// Sizes from reference: B=4, T=2048, C=768, E=8, K=2, hidden=4C=3072.
#define NTOK 8192
#define CD   768
#define FD   3072
#define NE   8
#define FC   512          // F-chunk width for fused GEMM1->GELU->GEMM2
#define XP   776          // LDS pitch (bf16 elems) for x tile: 1552B, 16B-aligned, bank-spread
#define HP   520          // LDS pitch (bf16 elems) for h tile: 1040B, 16B-aligned

typedef __attribute__((ext_vector_type(16))) __bf16 v16bf;
typedef __attribute__((ext_vector_type(8)))  float  v8f;

union V16U { v16bf v; uint4 q[2]; };

__device__ __forceinline__ unsigned short f2bf(float f) {
  unsigned u = __float_as_uint(f);
  unsigned r = u + 0x7fffu + ((u >> 16) & 1u);   // round-to-nearest-even
  return (unsigned short)(r >> 16);
}

__device__ __forceinline__ v8f zero8() {
  v8f z = {0.f,0.f,0.f,0.f,0.f,0.f,0.f,0.f};
  return z;
}

// B operand (K x 16, bf16): lanes 0-15 hold N=lane, K = k0..k0+15;
// lanes 16-31 hold N=lane-16, K = k0+16..k0+31. Caller passes the per-lane
// base pointer (already includes +16 K for the upper half-wave): 32 contiguous bytes.
__device__ __forceinline__ v16bf ldb_g(const unsigned short* p) {
  V16U u;
  u.q[0] = *(const uint4*)(p);
  u.q[1] = *(const uint4*)(p + 8);
  return u.v;
}

// A operand (16 x 32, bf16) from an LDS row-major tile with pitch `pitch`:
// lanes 0-15: row=lane,  K = k0..k0+15
// lanes 16-31: row=lane-16, K = k0+16..k0+31
__device__ __forceinline__ v16bf lda_lds(const unsigned short* base, int row,
                                         int pitch, int k0, int lane) {
  int off = row * pitch + k0 + ((lane >= 16) ? 16 : 0);
  V16U u;
  u.q[0] = *(const uint4*)(base + off);
  u.q[1] = *(const uint4*)(base + off + 8);
  return u.v;
}

// ---------------------------------------------------------------------------
// Gating: one wave per token. logits = (x+noise)@Wg, softmax, top-2, renorm.
// Softmax denominator cancels in top-2 renormalization.
// ---------------------------------------------------------------------------
__global__ void k_gate(const float* __restrict__ x, const float* __restrict__ noise,
                       const float* __restrict__ Wg, float* __restrict__ combine) {
  const int wave = threadIdx.x >> 5;
  const int lane = threadIdx.x & 31;
  const int tok  = blockIdx.x * 8 + wave;
  const float* xr = x     + (size_t)tok * CD;
  const float* nr = noise + (size_t)tok * CD;

  float acc[NE];
#pragma unroll
  for (int e = 0; e < NE; ++e) acc[e] = 0.f;

  for (int c = lane; c < CD; c += 32) {
    float xv = xr[c] + nr[c];
    const float4* w4 = (const float4*)(Wg + (size_t)c * NE);
    float4 wa = w4[0], wb = w4[1];
    acc[0] += xv * wa.x; acc[1] += xv * wa.y; acc[2] += xv * wa.z; acc[3] += xv * wa.w;
    acc[4] += xv * wb.x; acc[5] += xv * wb.y; acc[6] += xv * wb.z; acc[7] += xv * wb.w;
  }
#pragma unroll
  for (int e = 0; e < NE; ++e) {
#pragma unroll
    for (int m = 16; m >= 1; m >>= 1) acc[e] += __shfl_xor(acc[e], m, 32);
  }
  // all lanes hold all 8 logits now
  float lmax = acc[0];
#pragma unroll
  for (int e = 1; e < NE; ++e) lmax = fmaxf(lmax, acc[e]);
  float p[NE];
#pragma unroll
  for (int e = 0; e < NE; ++e) p[e] = expf(acc[e] - lmax);

  int i1 = 0; float p1 = p[0];
#pragma unroll
  for (int e = 1; e < NE; ++e) { if (p[e] > p1) { p1 = p[e]; i1 = e; } }
  int i2 = -1; float p2 = -1.f;
#pragma unroll
  for (int e = 0; e < NE; ++e) { if (e != i1 && p[e] > p2) { p2 = p[e]; i2 = e; } }
  float inv = 1.f / (p1 + p2);
  if (lane < NE) {
    float v = (lane == i1) ? p1 * inv : ((lane == i2) ? p2 * inv : 0.f);
    combine[(size_t)tok * NE + lane] = v;
  }
}

// ---------------------------------------------------------------------------
// x (fp32) -> bf16, 4 elems/thread
// ---------------------------------------------------------------------------
__global__ void k_cvtx(const float* __restrict__ in, unsigned short* __restrict__ out, int n) {
  int i = (blockIdx.x * blockDim.x + threadIdx.x) * 4;
  if (i + 3 < n) {
    float4 v = *(const float4*)(in + i);
    ushort4 o;
    o.x = f2bf(v.x); o.y = f2bf(v.y); o.z = f2bf(v.z); o.w = f2bf(v.w);
    *(ushort4*)(out + i) = o;
  }
}

// ---------------------------------------------------------------------------
// Tiled transpose + fp32->bf16: out[c][r] = bf16(in[r][c]) per expert (z-dim).
// 32x32 tile via LDS, coalesced reads and writes. Output stays L2-resident
// (75.5 MB bf16 weights << 192 MB L2) for the GEMM kernel.
// ---------------------------------------------------------------------------
__global__ void k_tconv(const float* __restrict__ in, unsigned short* __restrict__ out,
                        int R, int Cc) {
  __shared__ unsigned short tile[32][33];
  const size_t mat = (size_t)R * Cc;
  const float* ip = in + (size_t)blockIdx.z * mat;
  unsigned short* op = out + (size_t)blockIdx.z * mat;
  const int c0 = blockIdx.x * 32, r0 = blockIdx.y * 32;
  for (int j = threadIdx.y; j < 32; j += 8)
    tile[j][threadIdx.x] = f2bf(ip[(size_t)(r0 + j) * Cc + c0 + threadIdx.x]);
  __syncthreads();
  for (int j = threadIdx.y; j < 32; j += 8)
    op[(size_t)(c0 + j) * R + r0 + threadIdx.x] = tile[threadIdx.x][j];
}

// ---------------------------------------------------------------------------
// Fused MoE FFN. One block = 16 tokens, 8 waves. Wave w owns output columns
// [w*96, w*96+96) (6 WMMA col-tiles) and hidden columns [w*64, w*64+64)
// (4 WMMA col-tiles) of each 512-wide F chunk. Loops all 8 experts (dense,
// matching the reference's masked-dense math) so each output element has
// exactly one deterministic writer.
// ---------------------------------------------------------------------------
__global__ void __launch_bounds__(256)
k_ffn(const unsigned short* __restrict__ xb,     // [N][C] bf16
      const unsigned short* __restrict__ W1t,    // [E][F][C] bf16 (transposed)
      const unsigned short* __restrict__ W2t,    // [E][C][F] bf16 (transposed)
      const float* __restrict__ b1,              // [E][F]
      const float* __restrict__ b2,              // [E][C]
      const float* __restrict__ combine,         // [N][E]
      float* __restrict__ out)                   // [N][C]
{
  __shared__ unsigned short lds_x[16 * XP];  // 24.25 KB
  __shared__ unsigned short lds_h[16 * HP];  // 16.25 KB
  __shared__ float lds_cmb[16 * NE];

  const int tid  = threadIdx.x;
  const int wave = tid >> 5;
  const int lane = tid & 31;
  const int hl   = lane & 15;
  const int hi16 = (lane >= 16) ? 16 : 0;
  const int row0 = blockIdx.x * 16;

  { // stage x tile (16 x 768 bf16) into LDS, 16B vectors
    const uint4* src = (const uint4*)(xb + (size_t)row0 * CD);
    uint4* dst = (uint4*)lds_x;
    for (int idx = tid; idx < 16 * (CD / 8); idx += 256) {
      int r = idx / (CD / 8), cc = idx % (CD / 8);
      dst[r * (XP / 8) + cc] = src[(size_t)r * (CD / 8) + cc];
    }
  }
  if (tid < 16 * NE) lds_cmb[tid] = combine[(size_t)row0 * NE + tid];
  __syncthreads();

  v8f outacc[6];
#pragma unroll
  for (int t = 0; t < 6; ++t) outacc[t] = zero8();

  for (int e = 0; e < NE; ++e) {
    const unsigned short* W1e = W1t + (size_t)e * FD * CD;  // [f][c]
    const unsigned short* W2e = W2t + (size_t)e * CD * FD;  // [c][f]
    v8f yacc[6];
#pragma unroll
    for (int t = 0; t < 6; ++t) yacc[t] = zero8();

    for (int fc = 0; fc < FD / FC; ++fc) {
      const int fbase = fc * FC;

      // ---- GEMM1: h_chunk[16 x 64/wave] = x[16x768] @ W1[:, chunk] ----
      v8f hacc[4];
#pragma unroll
      for (int t = 0; t < 4; ++t) hacc[t] = zero8();
      for (int k0 = 0; k0 < CD; k0 += 32) {
        v16bf a = lda_lds(lds_x, hl, XP, k0, lane);
        const unsigned short* bp =
            W1e + (size_t)(fbase + wave * 64 + hl) * CD + k0 + hi16;
#pragma unroll
        for (int t = 0; t < 4; ++t) {
          v16bf b = ldb_g(bp + (size_t)t * 16 * CD);
          hacc[t] = __builtin_amdgcn_wmma_f32_16x16x32_bf16(
              false, a, false, b, (short)0, hacc[t], false, false);
        }
      }

      __syncthreads();  // prior chunk's GEMM2 readers done with lds_h
      // ---- bias + exact GELU, write bf16 h chunk to LDS ----
#pragma unroll
      for (int t = 0; t < 4; ++t) {
        int col = wave * 64 + t * 16 + hl;
        float b1v = b1[(size_t)e * FD + fbase + col];
#pragma unroll
        for (int r = 0; r < 8; ++r) {
          float hv = hacc[t][r] + b1v;
          hv = 0.5f * hv * (1.0f + erff(hv * 0.70710678118654752f));
          int rr = (lane < 16) ? r : (r + 8);   // D-matrix row mapping
          lds_h[rr * HP + col] = f2bf(hv);
        }
      }
      __syncthreads();

      // ---- GEMM2: y[16 x 96/wave] += h_chunk[16x512] @ W2[chunk, :] ----
      for (int k0 = 0; k0 < FC; k0 += 32) {
        v16bf a = lda_lds(lds_h, hl, HP, k0, lane);
        const unsigned short* bp =
            W2e + (size_t)(wave * 96 + hl) * FD + fbase + k0 + hi16;
#pragma unroll
        for (int t = 0; t < 6; ++t) {
          v16bf b = ldb_g(bp + (size_t)t * 16 * FD);
          yacc[t] = __builtin_amdgcn_wmma_f32_16x16x32_bf16(
              false, a, false, b, (short)0, yacc[t], false, false);
        }
      }
    }

    // ---- weighted combine: out += combine[n,e] * (y + b2[e]) ----
#pragma unroll
    for (int t = 0; t < 6; ++t) {
      int col = wave * 96 + t * 16 + hl;
      float b2v = b2[(size_t)e * CD + col];
#pragma unroll
      for (int r = 0; r < 8; ++r) {
        int rr = (lane < 16) ? r : (r + 8);
        float cm = lds_cmb[rr * NE + e];
        outacc[t][r] += cm * (yacc[t][r] + b2v);
      }
    }
  }

  // ---- store (each element written exactly once) ----
#pragma unroll
  for (int t = 0; t < 6; ++t) {
    int col = wave * 96 + t * 16 + hl;
#pragma unroll
    for (int r = 0; r < 8; ++r) {
      int rr = (lane < 16) ? r : (r + 8);
      out[(size_t)(row0 + rr) * CD + col] = outacc[t][r];
    }
  }
}

// ---------------------------------------------------------------------------
// Workspace layout (bytes, 256B aligned):
//   combine : [N][E] f32                       @ 0         (256 KB)
//   xb      : [N][C] bf16                      @ 262144    (12.0 MB)
//   W1t     : [E][F][C] bf16                   @ 12845056  (36 MB)
//   W2t     : [E][C][F] bf16                   @ 50593792  (36 MB)
// total ~84.3 MB
// ---------------------------------------------------------------------------
extern "C" void kernel_launch(void* const* d_in, const int* in_sizes, int n_in,
                              void* d_out, int out_size, void* d_ws, size_t ws_size,
                              hipStream_t stream) {
  (void)in_sizes; (void)n_in; (void)out_size; (void)ws_size;
  const float* x     = (const float*)d_in[0];
  const float* noise = (const float*)d_in[1];
  const float* Wg    = (const float*)d_in[2];
  const float* W1    = (const float*)d_in[3];
  const float* b1    = (const float*)d_in[4];
  const float* W2    = (const float*)d_in[5];
  const float* b2    = (const float*)d_in[6];
  float* out = (float*)d_out;

  char* ws = (char*)d_ws;
  float*          combine = (float*)(ws);
  unsigned short* xb      = (unsigned short*)(ws + 262144);
  unsigned short* W1t     = (unsigned short*)(ws + 12845056);
  unsigned short* W2t     = (unsigned short*)(ws + 50593792);

  k_gate<<<NTOK / 8, 256, 0, stream>>>(x, noise, Wg, combine);
  k_cvtx<<<(NTOK * CD) / 1024, 256, 0, stream>>>(x, xb, NTOK * CD);
  dim3 tb(32, 8);
  k_tconv<<<dim3(FD / 32, CD / 32, NE), tb, 0, stream>>>(W1, W1t, CD, FD);
  k_tconv<<<dim3(CD / 32, FD / 32, NE), tb, 0, stream>>>(W2, W2t, FD, CD);
  k_ffn<<<NTOK / 16, 256, 0, stream>>>(xb, W1t, W2t, b1, b2, combine, out);
}
